// ImplicitLinear_773094114038
// MI455X (gfx1250) — matrix operand order repaired and checked by reading, not compiled, not run on hardware
//
#include <hip/hip_runtime.h>
#include <hip/hip_bf16.h>

#define IN_F   4096
#define OUT_F  4096
#define RANK   256
#define N_TOK  8192
#define K_SST  205
#define NFFT   4096
#define LOG2N  12

typedef __attribute__((ext_vector_type(16))) __bf16 v16bf;
typedef __attribute__((ext_vector_type(8)))  float  v8f;

__device__ __forceinline__ unsigned short f2bf(float f) {
    unsigned int u = __float_as_uint(f);
    u += 0x7FFFu + ((u >> 16) & 1u);          // round-to-nearest-even
    return (unsigned short)(u >> 16);
}

// ---------------------------------------------------------------- zero W
__global__ void zero_f32(float* p, int n) {
    int i = blockIdx.x * blockDim.x + threadIdx.x;
    if (i < n) p[i] = 0.f;
}

// ---------------------------------------------------------------- COO scatter into W (fp32)
__global__ void scatter_coo(const float* __restrict__ vals,
                            const int* __restrict__ rows,
                            const int* __restrict__ cols,
                            float* __restrict__ W, int nnz) {
    int i = blockIdx.x * blockDim.x + threadIdx.x;
    if (i < nnz) atomicAdd(&W[(size_t)rows[i] * IN_F + cols[i]], vals[i]);
}

// ---------------------------------------------------------------- W = U@V + S  ->  bf16
#define TO 16
__global__ __launch_bounds__(256) void wbuild(const float* __restrict__ U,
                                              const float* __restrict__ V,
                                              const float* __restrict__ Ws,
                                              unsigned short* __restrict__ Wb) {
    const int i  = blockIdx.x * blockDim.x + threadIdx.x;  // column 0..4095
    const int o0 = blockIdx.y * TO;                        // output-row tile
    float acc[TO];
#pragma unroll
    for (int t = 0; t < TO; ++t) acc[t] = 0.f;
    for (int r = 0; r < RANK; ++r) {
        float v = V[(size_t)r * IN_F + i];                 // coalesced; V lives in L2 (4 MB)
#pragma unroll
        for (int t = 0; t < TO; ++t)
            acc[t] = fmaf(U[(size_t)(o0 + t) * RANK + r], v, acc[t]);  // uniform -> s_load
    }
#pragma unroll
    for (int t = 0; t < TO; ++t) {
        float w = acc[t] + Ws[(size_t)(o0 + t) * IN_F + i];
        Wb[(size_t)(o0 + t) * IN_F + i] = f2bf(w);
    }
}

// ---------------------------------------------------------------- SST: FFT -> top-k -> iFFT -> bf16
__global__ __launch_bounds__(256) void sst_kernel(const float* __restrict__ x,
                                                  unsigned short* __restrict__ xf) {
    __shared__ float2 buf[NFFT];    // 32 KB
    __shared__ float  msq[NFFT];    // 16 KB
    __shared__ int    wsum[8];
    const int tid  = threadIdx.x;
    const int lane = tid & 31, wid = tid >> 5;
    const float* xr = x + (size_t)blockIdx.x * IN_F;

    for (int i = tid; i < NFFT; i += 256) buf[i] = make_float2(xr[i], 0.f);
    __syncthreads();

    // forward FFT: radix-2 DIF (natural in -> bit-reversed out; no permute needed)
    for (int stage = LOG2N - 1; stage >= 0; --stage) {
        const int   half = 1 << stage;
        const float c    = -3.14159265358979323846f / (float)half;  // -2*pi/len
        for (int t = tid; t < NFFT / 2; t += 256) {
            int j  = t & (half - 1);
            int i0 = ((t >> stage) << (stage + 1)) + j;
            int i1 = i0 + half;
            float2 u = buf[i0], v = buf[i1];
            float s, co; sincosf(c * (float)j, &s, &co);
            float2 d = make_float2(u.x - v.x, u.y - v.y);
            buf[i0]  = make_float2(u.x + v.x, u.y + v.y);
            buf[i1]  = make_float2(d.x * co - d.y * s, d.x * s + d.y * co);
        }
        __syncthreads();
    }

    for (int i = tid; i < NFFT; i += 256) {
        float2 v = buf[i];
        msq[i] = v.x * v.x + v.y * v.y;
    }
    __syncthreads();

    // exact k-th largest |X|^2: binary search on IEEE bit pattern (monotone for >=0)
    unsigned int cur = 0;
    for (int bit = 30; bit >= 0; --bit) {
        unsigned int cand = cur | (1u << bit);
        float tval = __uint_as_float(cand);
        int cnt = 0;
        for (int i = tid; i < NFFT; i += 256) cnt += (msq[i] >= tval) ? 1 : 0;
        for (int off = 16; off > 0; off >>= 1) cnt += __shfl_down(cnt, off, 32);
        if (lane == 0) wsum[wid] = cnt;
        __syncthreads();
        int total = 0;
#pragma unroll
        for (int w = 0; w < 8; ++w) total += wsum[w];
        if (total >= K_SST) cur = cand;    // uniform across block
        __syncthreads();
    }
    const float thresh = __uint_as_float(cur);

    // filter (keep msq >= thresh, matching mags >= kth) + conjugate for inverse trick
    for (int i = tid; i < NFFT; i += 256) {
        float2 v = buf[i];
        if (msq[i] < thresh) { v.x = 0.f; v.y = 0.f; }
        v.y = -v.y;
        buf[i] = v;
    }
    __syncthreads();

    // inverse: radix-2 DIT on conj data (bit-reversed in -> natural out); Re(fft(conj X))/N
    for (int stage = 0; stage < LOG2N; ++stage) {
        const int   half = 1 << stage;
        const float c    = -3.14159265358979323846f / (float)half;
        for (int t = tid; t < NFFT / 2; t += 256) {
            int j  = t & (half - 1);
            int i0 = ((t >> stage) << (stage + 1)) + j;
            int i1 = i0 + half;
            float2 u = buf[i0], v = buf[i1];
            float s, co; sincosf(c * (float)j, &s, &co);
            float2 vw = make_float2(v.x * co - v.y * s, v.x * s + v.y * co);
            buf[i0] = make_float2(u.x + vw.x, u.y + vw.y);
            buf[i1] = make_float2(u.x - vw.x, u.y - vw.y);
        }
        __syncthreads();
    }

    unsigned short* out = xf + (size_t)blockIdx.x * IN_F;
    const float invn = 1.0f / (float)NFFT;
    for (int i = tid; i < NFFT; i += 256) out[i] = f2bf(buf[i].x * invn);
}

// ---------------------------------------------------------------- out = xf @ W^T + bias  (bf16 WMMA)
// Per-wave tile: 32 (tokens) x 64 (features) = 2x4 WMMA tiles.
// All 12 b128 loads of a k-step are issued before the 8 WMMAs so the
// compiler can emit one load clause and staggered s_wait_loadcnt.
union FragAB { uint4 u[2]; v16bf v; };

__global__ __launch_bounds__(256) void gemm_wmma(const unsigned short* __restrict__ A,
                                                 const unsigned short* __restrict__ B,
                                                 const float* __restrict__ bias,
                                                 float* __restrict__ out) {
    const int tid   = threadIdx.x;
    const int lane  = tid & 31;
    const int wv    = tid >> 5;                       // 8 waves: 2 (m) x 4 (n)
    const int m16   = lane & 15;
    const int hi    = lane >> 4;
    const int mbase = (blockIdx.x * 2 + (wv >> 2)) * 32;
    const int nbase = blockIdx.y * 256 + (wv & 3) * 64;

    v8f acc[8];                                       // [m2][t] -> acc[m2*4+t]
#pragma unroll
    for (int t = 0; t < 8; ++t)
#pragma unroll
        for (int e = 0; e < 8; ++e) acc[t][e] = 0.f;

    // A frag: elems 0..15 <- K + hi*16 .. +15
    const unsigned short* arow0 = A + (size_t)(mbase + m16) * IN_F + hi * 16;
    const unsigned short* arow1 = arow0 + (size_t)16 * IN_F;
    // B frag (row of W): elems 0..15 <- K + hi*16 .. +15
    const unsigned short* brow0 = B + (size_t)(nbase + m16) * IN_F + hi * 16;
    const unsigned short* brow1 = brow0 + (size_t)16 * IN_F;
    const unsigned short* brow2 = brow0 + (size_t)32 * IN_F;
    const unsigned short* brow3 = brow0 + (size_t)48 * IN_F;

    for (int k = 0; k < IN_F; k += 32) {
        FragAB fa0, fa1, fb0, fb1, fb2, fb3;
        fa0.u[0] = *(const uint4*)(arow0 + k);
        fa0.u[1] = *(const uint4*)(arow0 + k + 8);
        fa1.u[0] = *(const uint4*)(arow1 + k);
        fa1.u[1] = *(const uint4*)(arow1 + k + 8);
        fb0.u[0] = *(const uint4*)(brow0 + k);
        fb0.u[1] = *(const uint4*)(brow0 + k + 8);
        fb1.u[0] = *(const uint4*)(brow1 + k);
        fb1.u[1] = *(const uint4*)(brow1 + k + 8);
        fb2.u[0] = *(const uint4*)(brow2 + k);
        fb2.u[1] = *(const uint4*)(brow2 + k + 8);
        fb3.u[0] = *(const uint4*)(brow3 + k);
        fb3.u[1] = *(const uint4*)(brow3 + k + 8);

        acc[0] = __builtin_amdgcn_wmma_f32_16x16x32_bf16(
            false, fa0.v, false, fb0.v, (short)0, acc[0], false, false);
        acc[4] = __builtin_amdgcn_wmma_f32_16x16x32_bf16(
            false, fa1.v, false, fb0.v, (short)0, acc[4], false, false);
        acc[1] = __builtin_amdgcn_wmma_f32_16x16x32_bf16(
            false, fa0.v, false, fb1.v, (short)0, acc[1], false, false);
        acc[5] = __builtin_amdgcn_wmma_f32_16x16x32_bf16(
            false, fa1.v, false, fb1.v, (short)0, acc[5], false, false);
        acc[2] = __builtin_amdgcn_wmma_f32_16x16x32_bf16(
            false, fa0.v, false, fb2.v, (short)0, acc[2], false, false);
        acc[6] = __builtin_amdgcn_wmma_f32_16x16x32_bf16(
            false, fa1.v, false, fb2.v, (short)0, acc[6], false, false);
        acc[3] = __builtin_amdgcn_wmma_f32_16x16x32_bf16(
            false, fa0.v, false, fb3.v, (short)0, acc[3], false, false);
        acc[7] = __builtin_amdgcn_wmma_f32_16x16x32_bf16(
            false, fa1.v, false, fb3.v, (short)0, acc[7], false, false);
    }

#pragma unroll
    for (int t = 0; t < 4; ++t) {
        const int   n  = nbase + t * 16 + m16;
        const float bv = bias[n];
#pragma unroll
        for (int r = 0; r < 8; ++r) {
            const int m0 = mbase + r + 8 * hi;
            out[(size_t)m0 * OUT_F + n]        = acc[t][r]     + bv;
            out[(size_t)(m0 + 16) * OUT_F + n] = acc[4 + t][r] + bv;
        }
    }
}

// ---------------------------------------------------------------- launch
extern "C" void kernel_launch(void* const* d_in, const int* in_sizes, int n_in,
                              void* d_out, int out_size, void* d_ws, size_t ws_size,
                              hipStream_t stream) {
    const float* x    = (const float*)d_in[0];
    const float* U    = (const float*)d_in[1];
    const float* V    = (const float*)d_in[2];
    const float* sv   = (const float*)d_in[3];
    const int*   sr   = (const int*)d_in[4];
    const int*   sc   = (const int*)d_in[5];
    const float* bias = (const float*)d_in[6];
    float* out = (float*)d_out;
    const int nnz = in_sizes[3];

    char* ws = (char*)d_ws;
    unsigned short* xf = (unsigned short*)ws;                                   // 64 MB bf16 x_f
    float*          Wf = (float*)(ws + (size_t)N_TOK * IN_F * 2);               // 64 MB fp32 W
    unsigned short* Wb = (unsigned short*)(ws + (size_t)N_TOK * IN_F * 2
                                              + (size_t)OUT_F * IN_F * 4);      // 32 MB bf16 W

    const int nW = OUT_F * IN_F;
    zero_f32<<<(nW + 255) / 256, 256, 0, stream>>>(Wf, nW);
    scatter_coo<<<(nnz + 255) / 256, 256, 0, stream>>>(sv, sr, sc, Wf, nnz);
    wbuild<<<dim3(IN_F / 256, OUT_F / TO), 256, 0, stream>>>(U, V, Wf, Wb);
    sst_kernel<<<N_TOK, 256, 0, stream>>>(x, xf);
    gemm_wmma<<<dim3(N_TOK / 64, OUT_F / 256), 256, 0, stream>>>(xf, Wb, bias, out);
}
